// W8A8OF16LinearDynamicInputScale_37684043055693
// MI455X (gfx1250) — 8-bit matrix operand order repaired and checked by reading, compiled, not run on hardware
//
#include <hip/hip_runtime.h>

typedef int v8i __attribute__((ext_vector_type(8)));
typedef unsigned int v4u __attribute__((ext_vector_type(4)));
typedef int v4i __attribute__((ext_vector_type(4)));

#define Msz 8192
#define Nsz 4096
#define Ksz 4096
#define BM 128
#define BN 128
#define BK 64
#define LDSS 80

#if __has_builtin(__builtin_amdgcn_tensor_load_to_lds) && __has_builtin(__builtin_amdgcn_s_wait_tensorcnt)
#define USE_TDM 1
#else
#define USE_TDM 0
#endif

__global__ __launch_bounds__(256) void pack_i32_to_i8(const int4* __restrict__ src,
                                                      unsigned int* __restrict__ dst,
                                                      int count) {
    int i = blockIdx.x * 256 + threadIdx.x;
    if (i < count) {
        int4 v = src[i];
        dst[i] = (v.x & 0xFF) | ((v.y & 0xFF) << 8) | ((v.z & 0xFF) << 16) | (v.w << 24);
    }
}

#if USE_TDM
__device__ __forceinline__ void tdm_tile(const signed char* gptr, unsigned lds_off,
                                         unsigned tensor_rows) {
    unsigned long long ga = (unsigned long long)(const void*)gptr;
    v4u g0;
    g0[0] = 1u;
    g0[1] = lds_off;
    g0[2] = (unsigned)(ga & 0xFFFFFFFFu);
    g0[3] = (unsigned)((ga >> 32) & 0x01FFFFFFu)
            | 0x80000000u;
    v8i g1;
    g1[0] = (int)((1u << 20)
                | (3u << 22)
                | (3u << 25));
    g1[1] = (int)((unsigned)Ksz << 16);
    g1[2] = (int)(tensor_rows << 16);
    g1[3] = (int)(64u << 16);
    g1[4] = 128;
    g1[5] = Ksz;
    g1[6] = 0;
    g1[7] = 0;
    v4i g2; g2[0] = 1; g2[1] = 0; g2[2] = 0; g2[3] = 0;
    v4i g3; g3[0] = 0; g3[1] = 0; g3[2] = 0; g3[3] = 0;
#if __has_include(<hip/amd_detail/amd_gfx1250_TDM.h>)
    v8i g4 = {};
    __builtin_amdgcn_tensor_load_to_lds(g0, g1, g2, g3, g4, 0);
#else
    __builtin_amdgcn_tensor_load_to_lds(g0, g1, g2, g3, 0);
#endif
}
#endif

__global__ __launch_bounds__(256) void w8a8_wmma_gemm(
    const signed char* __restrict__ Ap,
    const signed char* __restrict__ Bp,
    const float* __restrict__ s_in,
    const float* __restrict__ s_w,
    const float* __restrict__ bias,
    _Float16* __restrict__ out)
{
    __shared__ __align__(16) unsigned char lds[2][(BM + BN) * LDSS];

    const int tid  = threadIdx.x;
    const int lane = tid & 31;
    const int wave = tid >> 5;
    const int wm   = wave & 3;
    const int wn   = wave >> 2;
    const int lo   = lane & 15;
    const int hi   = lane >> 4;

    const int m_base = blockIdx.y * BM;
    const int n_base = blockIdx.x * BN;

    v8i acc[2][4];
#pragma unroll
    for (int mt = 0; mt < 2; ++mt)
#pragma unroll
        for (int nt = 0; nt < 4; ++nt)
#pragma unroll
            for (int v = 0; v < 8; ++v) acc[mt][nt][v] = 0;

#if USE_TDM
    const unsigned ldsA[2] = { (unsigned)(size_t)&lds[0][0],
                               (unsigned)(size_t)&lds[1][0] };
    const unsigned ldsB[2] = { ldsA[0] + BM * LDSS, ldsA[1] + BM * LDSS };

    if (wave == 0) {
        tdm_tile(Ap + (size_t)m_base * Ksz, ldsA[0], (unsigned)Msz);
        tdm_tile(Bp + (size_t)n_base * Ksz, ldsB[0], (unsigned)Nsz);
        __builtin_amdgcn_s_wait_tensorcnt(0);
    }
    __syncthreads();
#else
    const int c0 = tid, c1 = tid + 256;
    const int ra_row0 = c0 >> 2, ra_off0 = (c0 & 3) * 16;
    const int ra_row1 = c1 >> 2, ra_off1 = (c1 & 3) * 16;
    int4 ra0, ra1, rb0, rb1;
    auto load_global = [&](int k0) {
        ra0 = *(const int4*)(Ap + (size_t)(m_base + ra_row0) * Ksz + k0 + ra_off0);
        ra1 = *(const int4*)(Ap + (size_t)(m_base + ra_row1) * Ksz + k0 + ra_off1);
        rb0 = *(const int4*)(Bp + (size_t)(n_base + ra_row0) * Ksz + k0 + ra_off0);
        rb1 = *(const int4*)(Bp + (size_t)(n_base + ra_row1) * Ksz + k0 + ra_off1);
    };
    auto store_lds = [&](int buf) {
        unsigned char* smA = lds[buf];
        unsigned char* smB = lds[buf] + BM * LDSS;
        *(int4*)(smA + ra_row0 * LDSS + ra_off0) = ra0;
        *(int4*)(smA + ra_row1 * LDSS + ra_off1) = ra1;
        *(int4*)(smB + ra_row0 * LDSS + ra_off0) = rb0;
        *(int4*)(smB + ra_row1 * LDSS + ra_off1) = rb1;
    };
    load_global(0);
    store_lds(0);
    __syncthreads();
#endif

    int it = 0;
    for (int k0 = 0; k0 < Ksz; k0 += BK, ++it) {
        const int cur = it & 1;
        const bool more = (k0 + BK) < Ksz;

#if USE_TDM
        if (more && wave == 0) {
            tdm_tile(Ap + (size_t)m_base * Ksz + (k0 + BK), ldsA[cur ^ 1], (unsigned)Msz);
            tdm_tile(Bp + (size_t)n_base * Ksz + (k0 + BK), ldsB[cur ^ 1], (unsigned)Nsz);
        }
#else
        if (more) load_global(k0 + BK);
#endif

        const unsigned char* smA = lds[cur];
        const unsigned char* smB = lds[cur] + BM * LDSS;

        v8i afrag[2];
#pragma unroll
        for (int mt = 0; mt < 2; ++mt) {
            const unsigned char* pa = smA + (wm * 32 + mt * 16 + lo) * LDSS + hi * 16;
#pragma unroll
            for (int j = 0; j < 4; ++j) {
                int2 q = *(const int2*)(pa + (j >> 1) * 32 + (j & 1) * 8);
                afrag[mt][2 * j]     = q.x;
                afrag[mt][2 * j + 1] = q.y;
            }
        }

        v8i bfrag[4];
#pragma unroll
        for (int nt = 0; nt < 4; ++nt) {
            const unsigned char* pb = smB + (wn * 64 + nt * 16 + lo) * LDSS + hi * 16;
            int4 q0 = *(const int4*)(pb);
            int4 q1 = *(const int4*)(pb + 32);
            bfrag[nt][0] = q0.x; bfrag[nt][1] = q0.y; bfrag[nt][2] = q0.z; bfrag[nt][3] = q0.w;
            bfrag[nt][4] = q1.x; bfrag[nt][5] = q1.y; bfrag[nt][6] = q1.z; bfrag[nt][7] = q1.w;
        }

#pragma unroll
        for (int mt = 0; mt < 2; ++mt)
#pragma unroll
            for (int nt = 0; nt < 4; ++nt)
                acc[mt][nt] = __builtin_amdgcn_wmma_i32_16x16x64_iu8(
                    true, afrag[mt], true, bfrag[nt], acc[mt][nt], false, false);

#if USE_TDM
        if (more && wave == 0) __builtin_amdgcn_s_wait_tensorcnt(0);
        __syncthreads();
#else
        if (more) store_lds(cur ^ 1);
        __syncthreads();
#endif
    }

    float si[2][8];
#pragma unroll
    for (int mt = 0; mt < 2; ++mt)
#pragma unroll
        for (int v = 0; v < 8; ++v)
            si[mt][v] = s_in[m_base + wm * 32 + mt * 16 + hi * 8 + v];

#pragma unroll
    for (int nt = 0; nt < 4; ++nt) {
        const int n = n_base + wn * 64 + nt * 16 + lo;
        const float sw = s_w[n];
        const float bb = bias[n];
#pragma unroll
        for (int mt = 0; mt < 2; ++mt) {
            const int mrow = m_base + wm * 32 + mt * 16 + hi * 8;
#pragma unroll
            for (int v = 0; v < 8; ++v) {
                float f = (float)acc[mt][nt][v] * si[mt][v] * sw + bb;
                out[(size_t)(mrow + v) * Nsz + n] = (_Float16)f;
            }
        }
    }
}

extern "C" void kernel_launch(void* const* d_in, const int* in_sizes, int n_in,
                              void* d_out, int out_size, void* d_ws, size_t ws_size,
                              hipStream_t stream) {
    const int*   x    = (const int*)d_in[0];
    const int*   w    = (const int*)d_in[1];
    const float* s_in = (const float*)d_in[2];
    const float* s_w  = (const float*)d_in[3];
    const float* bias = (const float*)d_in[4];
    _Float16*    out  = (_Float16*)d_out;

    signed char* xp = (signed char*)d_ws;
    signed char* wp = xp + (size_t)Msz * Ksz;
    (void)ws_size; (void)in_sizes; (void)n_in; (void)out_size;

    {
        int cx = (Msz * Ksz) / 4;
        int cw = (Nsz * Ksz) / 4;
        pack_i32_to_i8<<<(cx + 255) / 256, 256, 0, stream>>>(
            (const int4*)x, (unsigned int*)xp, cx);
        pack_i32_to_i8<<<(cw + 255) / 256, 256, 0, stream>>>(
            (const int4*)w, (unsigned int*)wp, cw);
    }

    dim3 grid(Nsz / BN, Msz / BM);
    w8a8_wmma_gemm<<<grid, 256, 0, stream>>>(xp, wp, s_in, s_w, bias, out);
}
